// Receiver_54168127537658
// MI455X (gfx1250) — matrix operand order repaired and checked by reading, not compiled, not run on hardware
//
#include <hip/hip_runtime.h>
#include <hip/hip_bf16.h>

// ---------------------------------------------------------------------------
// RGCN x2 + logits + gumbel-max sample + entropy for MI455X (gfx1250).
// Compute-bound on 9 GEMMs/layer of [32768 x 256] @ [256 x 256] (f32 data).
// Strategy: f32 aggregation (atomics), fold all relations + root into ONE
// GEMM per layer with K = 8*256 + 256 = 2304, run it through
// v_wmma_f32_16x16x32_bf16 (f32 accumulate), converting f32->bf16 while
// staging tiles into LDS. Weights pre-transposed to bf16 [O][K].
// Double-buffered K-loop with GLOBAL_LOAD_ASYNC_TO_LDS_B128 weight staging
// (ASYNCcnt / s_wait_asynccnt); builtin takes AS1/AS3 int4* pointers.
// ---------------------------------------------------------------------------

typedef __attribute__((ext_vector_type(16))) __bf16 v16bf;
typedef __attribute__((ext_vector_type(8)))  __bf16 v8bf;
typedef __attribute__((ext_vector_type(8)))  float  v8f;
typedef __attribute__((ext_vector_type(4)))  int    v4i;

#define Bsz 8
#define Nn  4096
#define Ff  256
#define Rr  8
#define Oo  256
#define GK  2304          // R*F + F
#define AK  2048          // R*F  (agg part of K)
#define Mm  32768         // B*N

#if __has_builtin(__builtin_amdgcn_global_load_async_to_lds_b128) && \
    __has_builtin(__builtin_amdgcn_s_wait_asynccnt)
#define ASYNC_LDS 1
#else
#define ASYNC_LDS 0
#endif

__device__ __forceinline__ void async_copy_b128(const void* g, void* l) {
#if ASYNC_LDS
    __builtin_amdgcn_global_load_async_to_lds_b128(
        (__attribute__((address_space(1))) v4i*)(void*)g,
        (__attribute__((address_space(3))) v4i*)l, 0, 0);
#endif
}

__device__ __forceinline__ void wait_async_lds() {
#if ASYNC_LDS
    __builtin_amdgcn_s_wait_asynccnt(0);
#endif
}

// ------------------------------- utility -----------------------------------

__global__ void zero_kernel(float* __restrict__ p, size_t n) {
    size_t i = ((size_t)blockIdx.x * blockDim.x + threadIdx.x) * 4;
    if (i < n) *(float4*)(p + i) = make_float4(0.f, 0.f, 0.f, 0.f);
}

__global__ void count_kernel(const int* __restrict__ ei, const int* __restrict__ et,
                             float* __restrict__ cnt, int E) {
    int e = blockIdx.x * 256 + threadIdx.x;
    if (e < E) {
        int d = ei[E + e];
        int r = et[e];
        atomicAdd(&cnt[d * Rr + r], 1.0f);
    }
}

__global__ void invert_kernel(float* __restrict__ cnt) {
    int i = blockIdx.x * 256 + threadIdx.x;   // N*R = 32768 total
    cnt[i] = 1.0f / fmaxf(cnt[i], 1.0f);
}

// WT[o][k] (bf16), k = r*256+f for k<2048, else root column (k-2048).
__global__ void prepw_kernel(const float* __restrict__ W, const float* __restrict__ root,
                             __bf16* __restrict__ WT) {
    int idx = blockIdx.x * 256 + threadIdx.x;     // Oo*GK = 589824
    if (idx >= Oo * GK) return;
    int o = idx / GK;
    int k = idx - o * GK;
    float v;
    if (k < AK) {
        int r = k >> 8, f = k & 255;
        v = W[((size_t)r * Ff + f) * Oo + o];
    } else {
        v = root[(size_t)(k - AK) * Oo + o];
    }
    WT[idx] = (__bf16)v;
}

// agg[(b*N+n)][r*256+f] += act[b][src][f] * norm ; one block per edge.
__global__ void aggregate_kernel(const float* __restrict__ act,
                                 const int* __restrict__ ei, const int* __restrict__ et,
                                 const float* __restrict__ norm,
                                 float* __restrict__ agg, int E) {
    int e = blockIdx.x;
    int f = threadIdx.x;
    int s = ei[e];
    int d = ei[E + e];
    int r = et[e];
    float w = norm[d * Rr + r];
#pragma unroll
    for (int b = 0; b < Bsz; ++b) {
        float v = act[((size_t)b * Nn + s) * Ff + f] * w;
        atomicAdd(&agg[((size_t)b * Nn + d) * (size_t)AK + (size_t)r * 256 + f], v);
    }
}

// ------------------------------- GEMM --------------------------------------
// A (f32, converted to bf16 in staging) = [agg | act] : [M=32768][K=2304]
// Bmat (bf16, pre-transposed) = WT[O=256][K=2304]
// out[m][o] = sum_k A[m][k]*W[k][o] + bias[o]  (optional ReLU)
// Block: 256 thr = 8 waves; tile M=64, Ncol=256; wave -> 16 x 128 strip.
// Double-buffered LDS; B staged via async-to-LDS DMA when available.

#define LDA_PITCH 40   // 32 bf16 + 8 pad
__global__ __launch_bounds__(256) void gemm_kernel(const float* __restrict__ agg,
                                                   const float* __restrict__ act,
                                                   const __bf16* __restrict__ WT,
                                                   const float* __restrict__ bias,
                                                   float* __restrict__ out, int relu) {
    __shared__ __bf16 lA[2][64 * LDA_PITCH];     // 2 x  5.0 KB
    __shared__ __bf16 lB[2][256 * LDA_PITCH];    // 2 x 20.0 KB

    const int tid  = threadIdx.x;
    const int wave = tid >> 5;
    const int lane = tid & 31;
    const int wm   = wave & 3;      // 4 M-subtiles of 16
    const int wn   = wave >> 2;     // 2 N-halves of 128
    const size_t m0 = (size_t)blockIdx.x * 64;

    const int r    = lane & 15;
    const int half = lane >> 4;

    v8f acc[8] = {};                // 8 x (16x16 f32) = 16 x 128 strip

    // stage tile (kc) into LDS buffer sbuf: A = sync convert, B = async DMA
    auto stage = [&](int sbuf, int kc) {
        {   // A tile 64x32 : thread -> (row, 8-elem quarter), f32 -> bf16
            int row = tid >> 2, q = tid & 3;
            const float* src;
            if (kc < AK) src = agg + (m0 + row) * (size_t)AK + kc + q * 8;
            else         src = act + (m0 + row) * (size_t)Ff + (kc - AK) + q * 8;
            float4 f0 = ((const float4*)src)[0];
            float4 f1 = ((const float4*)src)[1];
            v8bf a = { (__bf16)f0.x, (__bf16)f0.y, (__bf16)f0.z, (__bf16)f0.w,
                       (__bf16)f1.x, (__bf16)f1.y, (__bf16)f1.z, (__bf16)f1.w };
            *(v8bf*)&lA[sbuf][row * LDA_PITCH + q * 8] = a;
        }
        {   // B tile 256x32 : thread -> one O-row (64 bytes)
            const __bf16* g = WT + (size_t)tid * GK + kc;
#if ASYNC_LDS
            char* l = (char*)&lB[sbuf][tid * LDA_PITCH];
#pragma unroll
            for (int i = 0; i < 4; ++i)
                async_copy_b128((const char*)g + i * 16, l + i * 16);
#else
            const uint4* gp = (const uint4*)g;
            uint4* lp = (uint4*)&lB[sbuf][tid * LDA_PITCH];
            lp[0] = gp[0]; lp[1] = gp[1]; lp[2] = gp[2]; lp[3] = gp[3];
#endif
        }
    };

    stage(0, 0);
    wait_async_lds();
    __syncthreads();

    int buf = 0;
    for (int kc = 0; kc < GK; kc += 32) {
        if (kc + 32 < GK) stage(buf ^ 1, kc + 32);   // prefetch next chunk

        // ---- A fragment: row = wm*16 + (lane&15); K halves per ISA layout
        v8bf alo = *(const v8bf*)&lA[buf][(wm * 16 + r) * LDA_PITCH + half * 8];
        v8bf ahi = *(const v8bf*)&lA[buf][(wm * 16 + r) * LDA_PITCH + 16 + half * 8];
        v16bf af = __builtin_shufflevector(alo, ahi, 0,1,2,3,4,5,6,7,8,9,10,11,12,13,14,15);

#pragma unroll
        for (int nt = 0; nt < 8; ++nt) {
            int col = wn * 128 + nt * 16 + r;
            v8bf blo = *(const v8bf*)&lB[buf][col * LDA_PITCH + half * 8];
            v8bf bhi = *(const v8bf*)&lB[buf][col * LDA_PITCH + 16 + half * 8];
            v16bf bfr = __builtin_shufflevector(blo, bhi, 0,1,2,3,4,5,6,7,8,9,10,11,12,13,14,15);
            acc[nt] = __builtin_amdgcn_wmma_f32_16x16x32_bf16(
                false, af, false, bfr, (short)0, acc[nt], false, false);
        }

        wait_async_lds();     // next-chunk DMA landed in LDS
        __syncthreads();      // everyone done reading buf / writing buf^1
        buf ^= 1;
    }

    // ---- epilogue: C/D layout -> row = m0 + wm*16 + v + half*8, col as above
#pragma unroll
    for (int nt = 0; nt < 8; ++nt) {
        int col = wn * 128 + nt * 16 + r;
        float bv = bias[col];
#pragma unroll
        for (int v = 0; v < 8; ++v) {
            size_t row = m0 + wm * 16 + v + half * 8;
            float x = acc[nt][v] + bv;
            if (relu) x = fmaxf(x, 0.f);
            out[row * Oo + col] = x;
        }
    }
}

// ------------------------------ logits -------------------------------------
// one wave per (b,n) row: logits[b][n] = dot(emb[b][n][:], message[b][:])
__global__ void logits_kernel(const float* __restrict__ emb,
                              const float* __restrict__ msg,
                              float* __restrict__ logits) {
    int wave = threadIdx.x >> 5, lane = threadIdx.x & 31;
    size_t row = (size_t)blockIdx.x * 8 + wave;     // 0..32767
    int b = (int)(row >> 12);
    const float* e = emb + row * Oo;
    const float* m = msg + (size_t)b * Oo;
    float s = 0.f;
#pragma unroll
    for (int i = lane; i < Oo; i += 32) s += e[i] * m[i];
#pragma unroll
    for (int off = 16; off; off >>= 1) s += __shfl_xor(s, off, 32);
    if (lane == 0) logits[row] = s;
}

// ----------------------------- finalize ------------------------------------
__device__ inline float u01_hash(unsigned long long x) {
    x += 0x9E3779B97F4A7C15ull;
    x = (x ^ (x >> 30)) * 0xBF58476D1CE4E5B9ull;
    x = (x ^ (x >> 27)) * 0x94D049BB133111EBull;
    x ^= x >> 31;
    return ((float)(x >> 40) + 0.5f) * (1.0f / 16777216.0f);
}

// out: [0..8) sample (as float), [8..8+32768) logits, [8+32768..+8) entropy
__global__ void finalize_kernel(const float* __restrict__ logits, float* __restrict__ out) {
    const int b = blockIdx.x;
    const int tid = threadIdx.x;
    const float* l = logits + (size_t)b * Nn;
    __shared__ float sA[256];
    __shared__ float sB[256];
    __shared__ int   sI[256];

    float lmax = -3.4e38f, gmax = -3.4e38f;
    int gidx = 0;
    for (int n = tid; n < Nn; n += 256) {
        float v = l[n];
        out[8 + (size_t)b * Nn + n] = v;
        lmax = fmaxf(lmax, v);
        float u = u01_hash((unsigned long long)(b * Nn + n));
        float g = -logf(-logf(u + 1e-20f) + 1e-20f);
        float t = v + g;
        if (t > gmax) { gmax = t; gidx = n; }
    }
    sA[tid] = lmax; sB[tid] = gmax; sI[tid] = gidx;
    __syncthreads();
    for (int s = 128; s > 0; s >>= 1) {
        if (tid < s) {
            sA[tid] = fmaxf(sA[tid], sA[tid + s]);
            if (sB[tid + s] > sB[tid]) { sB[tid] = sB[tid + s]; sI[tid] = sI[tid + s]; }
        }
        __syncthreads();
    }
    float m = sA[0];
    if (tid == 0) out[b] = (float)sI[0];
    __syncthreads();

    float s0 = 0.f, s1 = 0.f;
    for (int n = tid; n < Nn; n += 256) {
        float v = l[n];
        float e = __expf(v - m);
        s0 += e; s1 += e * v;
    }
    sA[tid] = s0; sB[tid] = s1;
    __syncthreads();
    for (int s = 128; s > 0; s >>= 1) {
        if (tid < s) { sA[tid] += sA[tid + s]; sB[tid] += sB[tid + s]; }
        __syncthreads();
    }
    if (tid == 0) {
        float S0 = sA[0], S1 = sB[0];
        out[8 + (size_t)Bsz * Nn + b] = logf(S0) + m - S1 / S0;
    }
}

// ------------------------------ launch -------------------------------------

extern "C" void kernel_launch(void* const* d_in, const int* in_sizes, int n_in,
                              void* d_out, int out_size, void* d_ws, size_t ws_size,
                              hipStream_t stream) {
    const float* message = (const float*)d_in[0];
    const float* x       = (const float*)d_in[1];
    const float* W1      = (const float*)d_in[2];
    const float* root1   = (const float*)d_in[3];
    const float* b1      = (const float*)d_in[4];
    const float* W2      = (const float*)d_in[5];
    const float* root2   = (const float*)d_in[6];
    const float* b2      = (const float*)d_in[7];
    const int*   ei      = (const int*)d_in[8];
    const int*   et      = (const int*)d_in[9];
    const int E = in_sizes[9];

    // workspace layout
    char* ws = (char*)d_ws;
    float*  norm   = (float*)ws;                                        //   128 KB
    float*  agg    = (float*)(ws + (1u << 20));                         //   256 MB
    float*  h      = (float*)(ws + (1u << 20) + 268435456u);            //    32 MB
    float*  emb    = (float*)(ws + (1u << 20) + 268435456u + 33554432u);//    32 MB
    float*  logits = (float*)(ws + (1u << 20) + 268435456u + 67108864u);//   128 KB
    __bf16* WT1    = (__bf16*)(ws + (1u << 20) + 268435456u + 67239936u);
    __bf16* WT2    = WT1 + (size_t)Oo * GK;

    float* out = (float*)d_out;

    // degree normalization (shared by both layers)
    zero_kernel<<<32, 256, 0, stream>>>(norm, (size_t)Nn * Rr);
    count_kernel<<<(E + 255) / 256, 256, 0, stream>>>(ei, et, norm, E);
    invert_kernel<<<(Nn * Rr) / 256, 256, 0, stream>>>(norm);

    // weight prep (bf16, transposed, root folded into K)
    prepw_kernel<<<(Oo * GK) / 256, 256, 0, stream>>>(W1, root1, WT1);
    prepw_kernel<<<(Oo * GK) / 256, 256, 0, stream>>>(W2, root2, WT2);

    const size_t aggN = (size_t)Bsz * Nn * AK;   // 67,108,864 floats

    // ---- layer 1
    zero_kernel<<<(unsigned)(aggN / 4 / 256), 256, 0, stream>>>(agg, aggN);
    aggregate_kernel<<<E, 256, 0, stream>>>(x, ei, et, norm, agg, E);
    gemm_kernel<<<Mm / 64, 256, 0, stream>>>(agg, x, WT1, b1, h, 1);

    // ---- layer 2
    zero_kernel<<<(unsigned)(aggN / 4 / 256), 256, 0, stream>>>(agg, aggN);
    aggregate_kernel<<<E, 256, 0, stream>>>(h, ei, et, norm, agg, E);
    gemm_kernel<<<Mm / 64, 256, 0, stream>>>(agg, h, WT2, b2, emb, 0);

    // ---- readout
    logits_kernel<<<Mm / 8, 256, 0, stream>>>(emb, message, logits);
    finalize_kernel<<<Bsz, 256, 0, stream>>>(logits, out);
}
